// GAT_71588514889935
// MI455X (gfx1250) — matrix operand order repaired and checked by reading, not compiled, not run on hardware
//
#include <hip/hip_runtime.h>
#include <hip/hip_bf16.h>
#include <math.h>

#define NN 50000
#define NE 800000
#define NET (NN + NE)      // 850000 edges incl. self loops
#define FDIM 128
#define NHEAD 2
#define CH 64
#define NG 128
#define L1DIM 256
#define L2DIM 128
#define OUTDIM 10

typedef __attribute__((ext_vector_type(16))) _Float16 v16h;
typedef __attribute__((ext_vector_type(8)))  _Float16 v8h;
typedef __attribute__((ext_vector_type(8)))  float    v8f;

// ---- order-preserving float<->uint for atomicMax on signed floats ----
__device__ __forceinline__ unsigned f2ord(float f) {
  unsigned u = __float_as_uint(f);
  return (u & 0x80000000u) ? ~u : (u | 0x80000000u);
}
__device__ __forceinline__ float ord2f(unsigned u) {
  return __uint_as_float((u & 0x80000000u) ? (u & 0x7FFFFFFFu) : ~u);
}

// f32 -> f16 row-major convert (layer input)
__global__ void cvt_f16(const float* __restrict__ in, _Float16* __restrict__ out, int n) {
  int i = blockIdx.x * blockDim.x + threadIdx.x;
  if (i < n) out[i] = (_Float16)in[i];
}

// Pre-swizzle W [128,128] f32 into B-fragment layout, f16:
// Wswz[((kt*8 + ntile)*32 + lane)*16 + i] = W[kt*32 + (lane>>4)*16 + i][ntile*16 + (lane&15)]
__global__ void swz_w(const float* __restrict__ W, _Float16* __restrict__ Wswz) {
  int idx = blockIdx.x * blockDim.x + threadIdx.x;
  if (idx >= 4 * 8 * 32 * 16) return;
  int i    = idx & 15;
  int lane = (idx >> 4) & 31;
  int nt   = (idx >> 9) & 7;
  int kt   = idx >> 12;
  int K    = (kt << 5) + ((lane >> 4) << 4) + i;
  int Ncol = (nt << 4) + (lane & 15);
  Wswz[idx] = (_Float16)W[K * FDIM + Ncol];
}

// =====================================================================
// H = X16 @ Wswz  ([rows,128] x [128,128] -> [rows,128] f32).
// grid.x = rows/16, block = 256 (8 waves; wave w owns column tile w).
// X tile staged in LDS; B fragments pre-swizzled in global.
// =====================================================================
__global__ void __launch_bounds__(256) gat_gemm128(const _Float16* __restrict__ X16,
                                                   const _Float16* __restrict__ Wswz,
                                                   float* __restrict__ Hout) {
  __shared__ _Float16 xs[16 * FDIM];           // 4 KB tile
  const int tid  = threadIdx.x;
  const int wave = tid >> 5;
  const int lane = tid & 31;
  const int hw   = lane >> 4;
  const int r    = lane & 15;
  const int m0   = blockIdx.x << 4;
  {
    int row = tid >> 4;                        // 0..15
    int col = (tid & 15) << 3;                 // 0..120 step 8
    *(v8h*)&xs[row * FDIM + col] =
        *(const v8h*)&X16[(size_t)(m0 + row) * FDIM + col];
  }
  __syncthreads();

  v8f acc = {};
#pragma unroll
  for (int kt = 0; kt < 4; ++kt) {
    const int kb = kt << 5;
    // A (16x32 f16): a[0..15] = row r, K = kb+hw*16+(0..15)
    v8h a0 = *(const v8h*)&xs[r * FDIM + kb + (hw << 4)];
    v8h a1 = *(const v8h*)&xs[r * FDIM + kb + 8 + (hw << 4)];
    v16h a;
#pragma unroll
    for (int i = 0; i < 8; ++i) { a[i] = a0[i]; a[8 + i] = a1[i]; }
    v16h b = *(const v16h*)&Wswz[(size_t)(((kt << 3) + wave) << 5 | lane) << 4];
    acc = __builtin_amdgcn_wmma_f32_16x16x32_f16(false, a, false, b,
                                                 (short)0, acc, false, false);
  }
  const int n0 = wave << 4;
#pragma unroll
  for (int v = 0; v < 8; ++v)
    Hout[(size_t)(m0 + v + (hw << 3)) * FDIM + n0 + r] = acc[v];
}

// alpha_s[n,h] = <H[n,h,:], a_src[h,:]> ; alpha_d likewise
__global__ void alpha_kernel(const float* __restrict__ H, const float* __restrict__ asrc,
                             const float* __restrict__ adst,
                             float* __restrict__ as_, float* __restrict__ ad_) {
  int i = blockIdx.x * blockDim.x + threadIdx.x;
  if (i >= NN * NHEAD) return;
  int n = i >> 1, h = i & 1;
  const float* hp = H + (size_t)n * FDIM + h * CH;
  float ss = 0.f, sd = 0.f;
#pragma unroll 4
  for (int c = 0; c < CH; ++c) {
    float v = hp[c];
    ss += v * asrc[h * CH + c];
    sd += v * adst[h * CH + c];
  }
  as_[i] = ss; ad_[i] = sd;
}

// out = bias broadcast; mmax = ordered(-inf)=0; ssum = 0
__global__ void layer_init(const float* __restrict__ bias, float* __restrict__ out,
                           unsigned* __restrict__ mmax, float* __restrict__ ssum) {
  int i = blockIdx.x * blockDim.x + threadIdx.x;
  if (i < NN * FDIM) out[i] = bias[i & (FDIM - 1)];
  if (i < NN * NHEAD) { mmax[i] = 0u; ssum[i] = 0.f; }
}

__global__ void edge_max(const int* __restrict__ ei, const float* __restrict__ as_,
                         const float* __restrict__ ad_, unsigned* __restrict__ mmax) {
  int e = blockIdx.x * blockDim.x + threadIdx.x;
  if (e >= NET) return;
  int s = (e < NE) ? ei[e] : (e - NE);
  int d = (e < NE) ? ei[NE + e] : (e - NE);
#pragma unroll
  for (int h = 0; h < NHEAD; ++h) {
    float t = as_[s * NHEAD + h] + ad_[d * NHEAD + h];
    t = (t > 0.f) ? t : 0.2f * t;               // leaky relu, slope 0.2
    atomicMax(&mmax[d * NHEAD + h], f2ord(t));
  }
}

__global__ void edge_exp(const int* __restrict__ ei, const float* __restrict__ as_,
                         const float* __restrict__ ad_, const unsigned* __restrict__ mmax,
                         float* __restrict__ exbuf, float* __restrict__ ssum) {
  int e = blockIdx.x * blockDim.x + threadIdx.x;
  if (e >= NET) return;
  int s = (e < NE) ? ei[e] : (e - NE);
  int d = (e < NE) ? ei[NE + e] : (e - NE);
#pragma unroll
  for (int h = 0; h < NHEAD; ++h) {
    float t = as_[s * NHEAD + h] + ad_[d * NHEAD + h];
    t = (t > 0.f) ? t : 0.2f * t;
    float ex = __expf(t - ord2f(mmax[d * NHEAD + h]));
    exbuf[e * NHEAD + h] = ex;
    atomicAdd(&ssum[d * NHEAD + h], ex);
  }
}

// one wave per edge; lane owns 4 channels; out[dst] += h[src] * a
__global__ void edge_aggr(const int* __restrict__ ei, const float* __restrict__ H,
                          const float* __restrict__ exbuf, const float* __restrict__ ssum,
                          float* __restrict__ out) {
  int w = blockIdx.x * (blockDim.x >> 5) + (threadIdx.x >> 5);
  if (w >= NET) return;
  int lane = threadIdx.x & 31;
  int s = (w < NE) ? ei[w] : (w - NE);
  int d = (w < NE) ? ei[NE + w] : (w - NE);
  int c0 = lane << 2;                 // 4 channels per lane, all in one head
  int h = c0 >> 6;
  float coef = exbuf[w * NHEAD + h] / (ssum[d * NHEAD + h] + 1e-16f);
  const float4 hv = *(const float4*)(H + (size_t)s * FDIM + c0);
  float* op = out + (size_t)d * FDIM + c0;
  atomicAdd(op + 0, hv.x * coef);
  atomicAdd(op + 1, hv.y * coef);
  atomicAdd(op + 2, hv.z * coef);
  atomicAdd(op + 3, hv.w * coef);
}

__global__ void relu_ip(float* __restrict__ p, int n) {
  int i = blockIdx.x * blockDim.x + threadIdx.x;
  if (i < n) p[i] = fmaxf(p[i], 0.f);
}

__global__ void pool_init(float* __restrict__ psum, float* __restrict__ pcnt) {
  int i = blockIdx.x * blockDim.x + threadIdx.x;
  if (i < NG * FDIM) psum[i] = 0.f;
  if (i < NG) pcnt[i] = 0.f;
}

__global__ void pool_accum(const float* __restrict__ h, const int* __restrict__ batch,
                           float* __restrict__ psum, float* __restrict__ pcnt) {
  int n = blockIdx.x;
  int c = threadIdx.x;
  int g = batch[n];
  atomicAdd(&psum[g * FDIM + c], h[(size_t)n * FDIM + c]);
  if (c == 0) atomicAdd(&pcnt[g], 1.0f);
}

__global__ void pool_fin(const float* __restrict__ psum, const float* __restrict__ pcnt,
                         float* __restrict__ g) {
  int i = blockIdx.x * blockDim.x + threadIdx.x;
  if (i < NG * FDIM) g[i] = psum[i] / fmaxf(pcnt[i >> 7], 1.0f);
}

__global__ void mlp_kernel(const float* __restrict__ in, const float* __restrict__ w,
                           const float* __restrict__ b, float* __restrict__ out,
                           int rows, int fin, int fout, int dorelu) {
  int idx = blockIdx.x * blockDim.x + threadIdx.x;
  if (idx >= rows * fout) return;
  int i = idx / fout, j = idx % fout;
  float s = b[j];
  for (int k = 0; k < fin; ++k) s += in[i * fin + k] * w[k * fout + j];
  out[idx] = dorelu ? fmaxf(s, 0.f) : s;
}

extern "C" void kernel_launch(void* const* d_in, const int* in_sizes, int n_in,
                              void* d_out, int out_size, void* d_ws, size_t ws_size,
                              hipStream_t stream) {
  const float* x      = (const float*)d_in[0];
  const int*   ei     = (const int*)d_in[1];     // [2,E]: src then dst
  const int*   batch  = (const int*)d_in[2];
  const float* lin1_w = (const float*)d_in[15];
  const float* lin1_b = (const float*)d_in[16];
  const float* lin2_w = (const float*)d_in[17];
  const float* lin2_b = (const float*)d_in[18];
  const float* out_w  = (const float*)d_in[19];
  const float* out_b  = (const float*)d_in[20];

  char* ws = (char*)d_ws;
  size_t off = 0;
  auto alloc = [&](size_t bytes) -> void* {
    void* p = ws + off;
    off = (off + bytes + 255) & ~(size_t)255;
    return p;
  };
  float*     Hbuf  = (float*)alloc((size_t)NN * FDIM * 4);
  float*     buf0  = (float*)alloc((size_t)NN * FDIM * 4);
  float*     buf1  = (float*)alloc((size_t)NN * FDIM * 4);
  _Float16*  X16   = (_Float16*)alloc((size_t)NN * FDIM * 2);
  _Float16*  Wswz  = (_Float16*)alloc((size_t)4 * 8 * 32 * 16 * 2);
  float*     as_   = (float*)alloc((size_t)NN * NHEAD * 4);
  float*     ad_   = (float*)alloc((size_t)NN * NHEAD * 4);
  unsigned*  mmax  = (unsigned*)alloc((size_t)NN * NHEAD * 4);
  float*     ssum  = (float*)alloc((size_t)NN * NHEAD * 4);
  float*     exbuf = (float*)alloc((size_t)NET * NHEAD * 4);
  float*     psum  = (float*)alloc((size_t)NG * FDIM * 4);
  float*     pcnt  = (float*)alloc((size_t)NG * 4);
  float*     gbuf  = (float*)alloc((size_t)NG * FDIM * 4);
  float*     t1    = (float*)alloc((size_t)NG * L1DIM * 4);
  float*     t2    = (float*)alloc((size_t)NG * L2DIM * 4);

  const float* cur = x;
  float* outs[2] = {buf0, buf1};
  for (int l = 0; l < 3; ++l) {
    const float* Wl  = (const float*)d_in[3 + 4 * l];
    const float* asr = (const float*)d_in[4 + 4 * l];
    const float* adt = (const float*)d_in[5 + 4 * l];
    const float* bl  = (const float*)d_in[6 + 4 * l];
    float* outp = outs[l & 1];

    cvt_f16<<<(NN * FDIM + 255) / 256, 256, 0, stream>>>(cur, X16, NN * FDIM);
    swz_w<<<(4 * 8 * 32 * 16 + 255) / 256, 256, 0, stream>>>(Wl, Wswz);
    gat_gemm128<<<NN / 16, 256, 0, stream>>>(X16, Wswz, Hbuf);
    alpha_kernel<<<(NN * NHEAD + 255) / 256, 256, 0, stream>>>(Hbuf, asr, adt, as_, ad_);
    layer_init<<<(NN * FDIM + 255) / 256, 256, 0, stream>>>(bl, outp, mmax, ssum);
    edge_max<<<(NET + 255) / 256, 256, 0, stream>>>(ei, as_, ad_, mmax);
    edge_exp<<<(NET + 255) / 256, 256, 0, stream>>>(ei, as_, ad_, mmax, exbuf, ssum);
    edge_aggr<<<NET / 8, 256, 0, stream>>>(ei, Hbuf, exbuf, ssum, outp);
    relu_ip<<<(NN * FDIM + 255) / 256, 256, 0, stream>>>(outp, NN * FDIM);
    cur = outp;
  }

  pool_init<<<(NG * FDIM + 255) / 256, 256, 0, stream>>>(psum, pcnt);
  pool_accum<<<NN, FDIM, 0, stream>>>(cur, batch, psum, pcnt);
  pool_fin<<<(NG * FDIM + 255) / 256, 256, 0, stream>>>(psum, pcnt, gbuf);

  mlp_kernel<<<(NG * L1DIM + 255) / 256, 256, 0, stream>>>(gbuf, lin1_w, lin1_b, t1, NG, FDIM, L1DIM, 1);
  mlp_kernel<<<(NG * L2DIM + 255) / 256, 256, 0, stream>>>(t1, lin2_w, lin2_b, t2, NG, L1DIM, L2DIM, 1);
  mlp_kernel<<<(NG * OUTDIM + 255) / 256, 256, 0, stream>>>(t2, out_w, out_b, (float*)d_out, NG, L2DIM, OUTDIM, 0);
}
